// FFM_15453292331638
// MI455X (gfx1250) — compile-verified, matrix operand order checked by reading, not run on hardware
//
#include <hip/hip_runtime.h>
#include <hip/hip_bf16.h>
#include <math.h>

#define BATCH   4096
#define FIELDS  22
#define FEATS   200000
#define DIMS    16
#define NPAIRS  231      // FIELDS*(FIELDS-1)/2
#define PADP    256      // pairs padded to multiple of 32 (2 pairs per WMMA K=32 slot)
#define WAVES   4        // waves (== samples) per block

typedef __attribute__((ext_vector_type(16))) _Float16 v16h;
typedef __attribute__((ext_vector_type(8)))  _Float16 v8h;
typedef __attribute__((ext_vector_type(8)))  float    v8f;

union Frag { v16h v; v8h h[2]; };

__global__ __launch_bounds__(WAVES * 32)
void ffm_fwd_kernel(const int* __restrict__ x,
                    const float* __restrict__ emb,
                    const float* __restrict__ linw,
                    const float* __restrict__ bias,
                    float* __restrict__ out)
{
    // per wave: Ua[256][16] + Ub[256][16] in f16 = 16 KB; 4 waves -> 64 KB
    __shared__ __align__(16) _Float16 sh[WAVES * 2 * PADP * DIMS];

    const int lane = threadIdx.x & 31;
    const int wave = threadIdx.x >> 5;
    const int b    = blockIdx.x * WAVES + wave;

    _Float16* Ua = &sh[wave * 2 * PADP * DIMS];
    _Float16* Ub = Ua + PADP * DIMS;

    // lane L (<22) holds this sample's feature id for field L
    const int xv = (lane < FIELDS) ? x[b * FIELDS + lane] : 0;

    // ---------------- gather phase ----------------
    // t enumerates 2*PADP rows: even t -> Ua[p] = E[i, x_j], odd t -> Ub[p] = E[j, x_i]
    for (int t = lane; t < 2 * PADP; t += 32) {
        const int p     = t >> 1;
        const int which = t & 1;
        const int pp    = (p < NPAIRS) ? p : (NPAIRS - 1);   // clamp for uniform decode

        // decode pair pp -> (i,j), i<j
        int i = 0, rem = pp;
        while (rem >= FIELDS - 1 - i) { rem -= FIELDS - 1 - i; ++i; }
        const int j = i + 1 + rem;

        const int fld  = which ? j : i;     // embedding field
        const int slot = which ? i : j;     // x slot supplying the feature id
        const int feat = __shfl(xv, slot, 32);  // all 32 lanes active here

        _Float16* dst = (which ? Ub : Ua) + p * DIMS;
        if (p < NPAIRS) {
            const float4* row = reinterpret_cast<const float4*>(
                emb + (size_t)(fld * FEATS + feat) * DIMS);
            const float4 f0 = row[0], f1 = row[1], f2 = row[2], f3 = row[3];
            const v8h h0 = { (_Float16)f0.x, (_Float16)f0.y, (_Float16)f0.z, (_Float16)f0.w,
                             (_Float16)f1.x, (_Float16)f1.y, (_Float16)f1.z, (_Float16)f1.w };
            const v8h h1 = { (_Float16)f2.x, (_Float16)f2.y, (_Float16)f2.z, (_Float16)f2.w,
                             (_Float16)f3.x, (_Float16)f3.y, (_Float16)f3.z, (_Float16)f3.w };
            *reinterpret_cast<v8h*>(dst)     = h0;
            *reinterpret_cast<v8h*>(dst + 8) = h1;
        } else {
            const v8h z = { (_Float16)0, (_Float16)0, (_Float16)0, (_Float16)0,
                            (_Float16)0, (_Float16)0, (_Float16)0, (_Float16)0 };
            *reinterpret_cast<v8h*>(dst)     = z;
            *reinterpret_cast<v8h*>(dst + 8) = z;
        }
    }

    __syncthreads();

    // ---------------- WMMA phase ----------------
    // CDNA5 16-bit A-matrix 16x32 layout: lane<16 holds M=lane, K=0..7 & 16..23;
    // lane>=16 holds M=lane-16, K=8..15 & 24..31.  B (32x16) mirrors with N.
    // A row m packs pairs (base+2m, base+2m+1); B col n packs (base+2n, base+2n+1),
    // so D[q][q] accumulates dot(pair 2q) + dot(pair 2q+1).
    const int idx = lane & 15;
    const int off = (lane >> 4) * 8;          // K sub-block start within a 16-dim row

    v8f c = {};
#pragma unroll
    for (int w = 0; w < PADP / 32; ++w) {     // 8 WMMAs, accumulate in one C
        const int r0 = w * 32 + 2 * idx;
        Frag a, bf;
        a.h[0]  = *reinterpret_cast<const v8h*>(Ua + (size_t)r0 * DIMS + off);
        a.h[1]  = *reinterpret_cast<const v8h*>(Ua + (size_t)(r0 + 1) * DIMS + off);
        bf.h[0] = *reinterpret_cast<const v8h*>(Ub + (size_t)r0 * DIMS + off);
        bf.h[1] = *reinterpret_cast<const v8h*>(Ub + (size_t)(r0 + 1) * DIMS + off);
        c = __builtin_amdgcn_wmma_f32_16x16x32_f16(
                /*neg_a=*/false, a.v, /*neg_b=*/false, bf.v,
                /*c_mod=*/(short)0, c, /*reuse_a=*/false, /*reuse_b=*/false);
    }

    // Diagonal of C: lanes 0-7 own D[e][e] in c[e] (e==lane);
    // lanes 24-31 own D[8+e][8+e] in c[e] (e==lane-24).
    float acc = 0.0f;
#pragma unroll
    for (int e = 0; e < 8; ++e) {
        const float d = c[e];
        acc += ((lane == e) || (lane == 24 + e)) ? d : 0.0f;
    }

    // fold in linear term (lanes 0..21) and bias (lane 0)
    if (lane < FIELDS) acc += linw[xv];
    if (lane == 0)     acc += bias[0];

    // wave32 reduction
#pragma unroll
    for (int s = 16; s > 0; s >>= 1)
        acc += __shfl_xor(acc, s, 32);

    if (lane == 0)
        out[b] = 1.0f / (1.0f + expf(-acc));
}

extern "C" void kernel_launch(void* const* d_in, const int* in_sizes, int n_in,
                              void* d_out, int out_size, void* d_ws, size_t ws_size,
                              hipStream_t stream) {
    (void)in_sizes; (void)n_in; (void)d_ws; (void)ws_size; (void)out_size;
    const int*   x    = (const int*)  d_in[0];
    const float* emb  = (const float*)d_in[1];
    const float* linw = (const float*)d_in[2];
    const float* bias = (const float*)d_in[3];
    float* out = (float*)d_out;

    dim3 grid(BATCH / WAVES);     // 1024 blocks
    dim3 block(WAVES * 32);       // 128 threads = 4 wave32s, one sample each
    ffm_fwd_kernel<<<grid, block, 0, stream>>>(x, emb, linw, bias, out);
}
